// Diffusion_86165633893151
// MI455X (gfx1250) — matrix operand order repaired and checked by reading, not compiled, not run on hardware
//
#include <hip/hip_runtime.h>

// bf16 WMMA vector types (probe-confirmed signature:
// (bool, v16bf, bool, v16bf, short, v8f, bool, bool))
typedef __bf16 v16bf __attribute__((ext_vector_type(16)));
typedef __bf16 v8bf  __attribute__((ext_vector_type(8)));
typedef float  v8f   __attribute__((ext_vector_type(8)));

#define KDIM 256   // eigenpairs
#define CDIM 128   // channels
#define RCHUNK 64  // N-rows staged per workgroup iteration in phase 1
#define EP 72      // padded LDS row stride (bf16 elems) for evecs-T chunk
#define XP 72      // padded LDS row stride for x-T chunk
#define KP (KDIM + 8)  // padded LDS row stride for B-T in phase 2
#define TILE_ELEMS (KDIM * CDIM)  // 32768 floats per [K,C] tile

// Phase 1: partials[g][K,C] = sum over this WG's chunks of evecs^T @ x.
// Register-resident fp32 accumulators across chunks; NO atomics.
// A-fragment loads are software-pipelined one mtile ahead.
__global__ __launch_bounds__(256) void spec_project(
    const float* __restrict__ evecs, const float* __restrict__ x,
    float* __restrict__ partials, int N, int chunks) {
  __shared__ __bf16 eT[KDIM][EP];  // evecs chunk, transposed: [col][row] 36864 B
  __shared__ __bf16 xT[CDIM][XP];  // x chunk,    transposed: [col][row] 18432 B
  const int tid = threadIdx.x;

  const int wave = tid >> 5;   // 8 waves -> 8 C-tiles (covers C=128)
  const int lane = tid & 31;
  const int half = lane >> 4;  // lane group 0-15 / 16-31
  const int lrow = lane & 15;
  const int ntile = wave;
  const int bcol = ntile * 16 + lrow;

  v8f acc[KDIM / 16];  // 16 x 8 = 128 VGPRs/lane of persistent accumulators
#pragma unroll
  for (int m = 0; m < KDIM / 16; ++m) acc[m] = (v8f){};

  for (int ch = blockIdx.x; ch < chunks; ch += gridDim.x) {
    const int n0 = ch * RCHUNK;
    __syncthreads();  // previous iteration's compute done before restaging
    // Stage evecs chunk (coalesced float4 loads -> bf16 -> LDS transposed)
    for (int v = tid; v < RCHUNK * (KDIM / 4); v += 256) {
      int row = v / (KDIM / 4);
      int c4  = v % (KDIM / 4);
      int gr = n0 + row;
      float4 f = {0.f, 0.f, 0.f, 0.f};
      if (gr < N) f = ((const float4*)(evecs + (size_t)gr * KDIM))[c4];
      int c = c4 * 4;
      eT[c + 0][row] = (__bf16)f.x;
      eT[c + 1][row] = (__bf16)f.y;
      eT[c + 2][row] = (__bf16)f.z;
      eT[c + 3][row] = (__bf16)f.w;
    }
    // Stage x chunk
    for (int v = tid; v < RCHUNK * (CDIM / 4); v += 256) {
      int row = v / (CDIM / 4);
      int c4  = v % (CDIM / 4);
      int gr = n0 + row;
      float4 f = {0.f, 0.f, 0.f, 0.f};
      if (gr < N) f = ((const float4*)(x + (size_t)gr * CDIM))[c4];
      int c = c4 * 4;
      xT[c + 0][row] = (__bf16)f.x;
      xT[c + 1][row] = (__bf16)f.y;
      xT[c + 2][row] = (__bf16)f.z;
      xT[c + 3][row] = (__bf16)f.w;
    }
    __syncthreads();

#pragma unroll
    for (int kk = 0; kk < RCHUNK; kk += 32) {
      // B 32x16 bf16 frag (shared by all 16 mtiles this kk):
      // lanes0-15 K {0..7,16..23}, lanes16-31 K {8..15,24..31}
      v8bf b0 = *(const v8bf*)&xT[bcol][kk + half * 8];
      v8bf b1 = *(const v8bf*)&xT[bcol][kk + 16 + half * 8];
      v16bf B;
#pragma unroll
      for (int i = 0; i < 8; ++i) { B[i] = b0[i]; B[i + 8] = b1[i]; }

      // A frag (16x32 bf16): lanes0-15 K {0..7,16..23}, lanes16-31 {8..15,24..31}
      // Software pipeline: prefetch next mtile's A pair before current WMMA.
      v8bf a0 = *(const v8bf*)&eT[lrow][kk + half * 8];
      v8bf a1 = *(const v8bf*)&eT[lrow][kk + 16 + half * 8];
#pragma unroll
      for (int mtile = 0; mtile < KDIM / 16; ++mtile) {
        v8bf p0 = a0, p1 = a1;
        if (mtile + 1 < KDIM / 16) {
          const int acol = (mtile + 1) * 16 + lrow;
          p0 = *(const v8bf*)&eT[acol][kk + half * 8];
          p1 = *(const v8bf*)&eT[acol][kk + 16 + half * 8];
        }
        v16bf A;
#pragma unroll
        for (int i = 0; i < 8; ++i) { A[i] = a0[i]; A[i + 8] = a1[i]; }
        acc[mtile] = __builtin_amdgcn_wmma_f32_16x16x32_bf16(
            false, A, false, B, (short)0, acc[mtile], false, false);
        a0 = p0; a1 = p1;
      }
    }
  }

  // Plain stores of this WG's full [K,C] partial (D: VGPR r -> M=r+8*half, N=lrow)
  float* my = partials + (size_t)blockIdx.x * TILE_ELEMS;
#pragma unroll
  for (int mtile = 0; mtile < KDIM / 16; ++mtile) {
#pragma unroll
    for (int r = 0; r < 8; ++r) {
      int krow = mtile * 16 + r + 8 * half;
      my[krow * CDIM + ntile * 16 + lrow] = acc[mtile][r];
    }
  }
}

// Sum G partials -> xs[K,C]
__global__ void reduce_partials(const float* __restrict__ partials,
                                float* __restrict__ xs, int G) {
  int i = blockIdx.x * blockDim.x + threadIdx.x;
  if (i < TILE_ELEMS) {
    float s = 0.f;
    for (int g = 0; g < G; ++g) s += partials[(size_t)g * TILE_ELEMS + i];
    xs[i] = s;
  }
}

// Phase 2: out[N,C] = evecs @ (exp(-evals*t) .* xs)
// All 8 B fragments hoisted into registers (mtile-invariant).
__global__ __launch_bounds__(128) void spec_synth(
    const float* __restrict__ evecs, const float* __restrict__ xs,
    const float* __restrict__ evals, const float* __restrict__ tptr,
    float* __restrict__ out, int N) {
  __shared__ __bf16 BsT[64][KP];  // scaled x_spec, transposed: [c-local][k] 33792 B
  const int tid = threadIdx.x;
  float t = tptr[0];
  t = t < 1e-8f ? 1e-8f : t;
  const int c0 = blockIdx.y * 64;  // this WG's 64-column half of C

  for (int v = tid; v < KDIM * 16; v += 128) {
    int k  = v >> 4;
    int c4 = v & 15;
    float coef = __expf(-evals[k] * t);
    float4 f = ((const float4*)(xs + k * CDIM + c0))[c4];
    int c = c4 * 4;
    BsT[c + 0][k] = (__bf16)(coef * f.x);
    BsT[c + 1][k] = (__bf16)(coef * f.y);
    BsT[c + 2][k] = (__bf16)(coef * f.z);
    BsT[c + 3][k] = (__bf16)(coef * f.w);
  }
  __syncthreads();

  const int wave = tid >> 5;  // 4 waves -> 4 local C-tiles
  const int lane = tid & 31;
  const int half = lane >> 4;
  const int lrow = lane & 15;
  const int bcol = wave * 16 + lrow;
  const int row0 = blockIdx.x * 128;

  // Hoist all 8 B fragments (covers full K=256) into registers: 64 VGPRs.
  v16bf Bfr[KDIM / 32];
#pragma unroll
  for (int j = 0; j < KDIM / 32; ++j) {
    const int k0 = j * 32;
    v8bf b0 = *(const v8bf*)&BsT[bcol][k0 + half * 8];
    v8bf b1 = *(const v8bf*)&BsT[bcol][k0 + 16 + half * 8];
#pragma unroll
    for (int i = 0; i < 8; ++i) { Bfr[j][i] = b0[i]; Bfr[j][i + 8] = b1[i]; }
  }

  for (int mtile = 0; mtile < 8; ++mtile) {
    int m0 = row0 + mtile * 16;
    int ar = m0 + lrow;
    if (ar >= N) ar = N - 1;  // clamp tail (stores are masked below)
    const float* arp = evecs + (size_t)ar * KDIM;
    v8f acc = {};
#pragma unroll
    for (int j = 0; j < KDIM / 32; ++j) {
      const int k0 = j * 32;
      // A frag straight from global: two 8-float contiguous runs per lane
      float4 fa0 = *(const float4*)(arp + k0 + 8 * half);
      float4 fa1 = *(const float4*)(arp + k0 + 8 * half + 4);
      float4 fa2 = *(const float4*)(arp + k0 + 16 + 8 * half);
      float4 fa3 = *(const float4*)(arp + k0 + 16 + 8 * half + 4);
      v16bf A;
      A[0] = (__bf16)fa0.x; A[1] = (__bf16)fa0.y; A[2] = (__bf16)fa0.z; A[3] = (__bf16)fa0.w;
      A[4] = (__bf16)fa1.x; A[5] = (__bf16)fa1.y; A[6] = (__bf16)fa1.z; A[7] = (__bf16)fa1.w;
      A[8] = (__bf16)fa2.x; A[9] = (__bf16)fa2.y; A[10] = (__bf16)fa2.z; A[11] = (__bf16)fa2.w;
      A[12] = (__bf16)fa3.x; A[13] = (__bf16)fa3.y; A[14] = (__bf16)fa3.z; A[15] = (__bf16)fa3.w;
      acc = __builtin_amdgcn_wmma_f32_16x16x32_bf16(
          false, A, false, Bfr[j], (short)0, acc, false, false);
    }
#pragma unroll
    for (int r = 0; r < 8; ++r) {
      int grow = m0 + r + 8 * half;
      if (grow < N) out[(size_t)grow * CDIM + c0 + wave * 16 + lrow] = acc[r];
    }
  }
}

extern "C" void kernel_launch(void* const* d_in, const int* in_sizes, int n_in,
                              void* d_out, int out_size, void* d_ws, size_t ws_size,
                              hipStream_t stream) {
  // setup_inputs order: x [N,C], evals [K], evecs [N,K], diffusion_time [1]
  const float* x     = (const float*)d_in[0];
  const float* evals = (const float*)d_in[1];
  const float* evecs = (const float*)d_in[2];
  const float* tptr  = (const float*)d_in[3];
  float* out = (float*)d_out;

  float* xs       = (float*)d_ws;        // final [K,C], 128 KB
  float* partials = xs + TILE_ELEMS;     // G x [K,C] partials

  const int N = in_sizes[0] / CDIM;
  const int chunks = (N + RCHUNK - 1) / RCHUNK;

  // Grid size for phase 1: fill the device but cap partial-buffer footprint.
  size_t avail = ws_size / sizeof(float);
  long gmax = (long)(avail > 2 * (size_t)TILE_ELEMS
                         ? (avail - TILE_ELEMS) / TILE_ELEMS : 1);
  int G = (int)(gmax < 1 ? 1 : gmax);
  if (G > 256) G = 256;
  if (G > chunks) G = chunks;

  spec_project<<<G, 256, 0, stream>>>(evecs, x, partials, N, chunks);
  reduce_partials<<<(TILE_ELEMS + 255) / 256, 256, 0, stream>>>(partials, xs, G);

  dim3 g2((N + 127) / 128, 2);
  spec_synth<<<g2, 128, 0, stream>>>(evecs, xs, evals, tptr, out, N);
}
